// Attention_38122129719388
// MI455X (gfx1250) — matrix operand order repaired and checked by reading, not compiled, not run on hardware
//
#include <hip/hip_runtime.h>

#define H    1024
#define NH   16
#define HD   64
#define S    2048
#define B    2
#define MROWS (B * S)          // 4096
#define EPS  1e-5f

typedef __attribute__((ext_vector_type(16))) __bf16 v16bf;
typedef __attribute__((ext_vector_type(8)))  float  v8f;
typedef __attribute__((ext_vector_type(4)))  unsigned int u32x4;

// ---------------------------------------------------------------------------
// WMMA helpers (CDNA5 wave32, V_WMMA_F32_16X16X32_BF16)
// ---------------------------------------------------------------------------
static __device__ __forceinline__ v8f wmma_bf16(v16bf a, v16bf b, v8f c) {
    // (neg_a, A, neg_b, B, c_mod, C, reuse_a, reuse_b)
    return __builtin_amdgcn_wmma_f32_16x16x32_bf16(false, a, false, b,
                                                   (short)0, c, false, false);
}

// CDNA5 async copy: global -> LDS, 16 bytes per lane, tracked by ASYNCcnt.
//   global_load_async_to_lds_b128 vdst(lds byte offset), v[addr:addr+1], off
// For a generic __shared__ pointer the low 32 bits are the as(3) LDS offset.
static __device__ __forceinline__ void async_copy_b128(unsigned int lds_off,
                                                       const __bf16* g) {
    asm volatile("global_load_async_to_lds_b128 %0, %1, off"
                 :: "v"(lds_off), "v"((unsigned long long)(size_t)g)
                 : "memory");
}

template <int N>
static __device__ __forceinline__ void wait_async() {
    asm volatile("s_wait_asynccnt %0" :: "i"(N) : "memory");
}

// A fragment: 16x32 (MxK), bf16, row-major source with leading dim ld (elems).
// Lane L (<16) holds row M=L, K=0..7 then 16..23 ; lane>=16: row M=L-16,
// K=8..15 then 24..31  -> two 16-byte contiguous runs per lane.
static __device__ __forceinline__ v16bf load_a_bf16(const __bf16* p, int ld, int lane) {
    const int row  = lane & 15;
    const int koff = (lane >> 4) << 3;            // 0 or 8
    const __bf16* r = p + (size_t)row * ld + koff;
    union { v16bf v; u32x4 u[2]; } f;
    f.u[0] = *(const u32x4*)(r);
    f.u[1] = *(const u32x4*)(r + 16);
    return f.v;
}

// B fragment: 32x16 (KxN). Source is row-major [N,K] (i.e. column n of B is a
// contiguous K-run). Lane L holds col N=L&15, K=(L>>4)*8 .. +7 then +16 .. +23.
static __device__ __forceinline__ v16bf load_b_bf16(const __bf16* p, int ld, int lane) {
    const int n    = lane & 15;
    const int koff = (lane >> 4) << 3;            // 0 or 8
    const __bf16* r = p + (size_t)n * ld + koff;
    union { v16bf v; u32x4 u[2]; } f;
    f.u[0] = *(const u32x4*)(r);
    f.u[1] = *(const u32x4*)(r + 16);
    return f.v;
}

// A fragment from fp32 source (convert to bf16 on the fly).
static __device__ __forceinline__ v16bf load_a_f32(const float* p, int ld, int lane) {
    const int row  = lane & 15;
    const int koff = (lane >> 4) << 3;
    const float* r = p + (size_t)row * ld + koff;
    v16bf v;
#pragma unroll
    for (int i = 0; i < 8; ++i) v[i] = (__bf16)r[i];
#pragma unroll
    for (int i = 0; i < 8; ++i) v[8 + i] = (__bf16)r[16 + i];
    return v;
}

// ---------------------------------------------------------------------------
// fp32 -> bf16 conversion
// ---------------------------------------------------------------------------
__global__ void cvt_bf16_kernel(const float* __restrict__ in,
                                __bf16* __restrict__ out, int n) {
    int i = blockIdx.x * 256 + threadIdx.x;
    if (i < n) out[i] = (__bf16)in[i];
}

// ---------------------------------------------------------------------------
// Generic NT GEMM:  C[M,N] = A[M,K] @ W[N,K]^T + bias[N]
//   MODE 0: fp32 row-major out, MODE 1: bf16 row-major out (scaled),
//   MODE 3: bf16 transposed-V out  Vt[b,h,d,s]
// Block: 256 thr (8 waves). Tile 128(M) x 64(N); wave w owns rows w*16..+15.
// B tile (64x32 bf16, shared by all 8 waves) is double-buffered in LDS via
// global_load_async_to_lds_b128 (one b128 per thread per k-step).
// ---------------------------------------------------------------------------
template <int MODE>
__global__ __launch_bounds__(256)
void gemm_bf16_kernel(const __bf16* __restrict__ A, const __bf16* __restrict__ W,
                      const float* __restrict__ bias, void* __restrict__ out,
                      int M, int N, int K, float scale) {
    __shared__ __bf16 Bt[2][64 * 32];
    const int tid  = threadIdx.x;
    const int lane = tid & 31;
    const int wave = tid >> 5;
    const int m0 = blockIdx.x * 128 + wave * 16;
    const int n0 = blockIdx.y * 64;

    // per-thread staging slot: one 16B chunk of the 4KB tile
    const int srow = tid >> 2;                 // 0..63  (N-row of tile)
    const int soff = (tid & 3) * 8;            // element offset inside 32-elem row
    const __bf16* gsrc = W + (size_t)(n0 + srow) * K + soff;
    const unsigned int l0 = (unsigned int)(size_t)&Bt[0][srow * 32 + soff];
    const unsigned int l1 = (unsigned int)(size_t)&Bt[1][srow * 32 + soff];

    async_copy_b128(l0, gsrc);                 // prologue: tile 0 -> buf 0

    v8f acc[4] = {};
    const int NIT = K / 32;
    for (int i = 0; i < NIT; ++i) {
        const int k = i * 32;
        if (i + 1 < NIT) {
            async_copy_b128((i & 1) ? l0 : l1, gsrc + (k + 32));
            wait_async<1>();                   // tile i landed (in-order)
        } else {
            wait_async<0>();
        }
        __syncthreads();                       // tile i visible to all waves

        v16bf a = load_a_bf16(A + (size_t)m0 * K + k, K, lane);
        const __bf16* bt = Bt[i & 1];
#pragma unroll
        for (int t = 0; t < 4; ++t)
            acc[t] = wmma_bf16(a, load_b_bf16(bt + t * 16 * 32, 32, lane), acc[t]);

        __syncthreads();                       // done reading before overwrite
    }

    const int rbase = m0 + ((lane >> 4) << 3);
    const int col16 = lane & 15;
#pragma unroll
    for (int t = 0; t < 4; ++t) {
        const int n = n0 + t * 16 + col16;
        const float bv = bias[n];
#pragma unroll
        for (int v = 0; v < 8; ++v) {
            const int m = rbase + v;
            const float val = (acc[t][v] + bv) * scale;
            if (MODE == 0) {
                ((float*)out)[(size_t)m * N + n] = val;
            } else if (MODE == 1) {
                ((__bf16*)out)[(size_t)m * N + n] = (__bf16)val;
            } else {  // MODE 3: Vt[b][h][d][s]
                const int bb = m >> 11, s = m & (S - 1);
                const int hh = n >> 6,  d = n & (HD - 1);
                ((__bf16*)out)[((size_t)((bb * NH + hh) * HD + d) << 11) + s] = (__bf16)val;
            }
        }
    }
}

// ---------------------------------------------------------------------------
// scores = (Q*0.125) @ K^T + mask ; softmax rows ; * head_mask -> probs (fp32)
// One block = 16 q-rows for one (b,h). 8 waves x 256 cols each (16 n-tiles).
// (No LDS staging: each wave consumes distinct K-rows; L2 covers reuse.)
// ---------------------------------------------------------------------------
__global__ __launch_bounds__(256)
void scores_softmax_kernel(const __bf16* __restrict__ Qb, const __bf16* __restrict__ Kb,
                           const float* __restrict__ mask, const float* __restrict__ head_mask,
                           float* __restrict__ probs) {
    const int qt = blockIdx.x & 127;          // S/16 tiles
    const int h  = (blockIdx.x >> 7) & (NH - 1);
    const int b  = blockIdx.x >> 11;
    const int lane = threadIdx.x & 31;
    const int wave = threadIdx.x >> 5;
    const int q0 = qt * 16;

    const __bf16* qp = Qb + ((size_t)(b * S + q0)) * H + h * HD;
    const v16bf qa0 = load_a_bf16(qp, H, lane);
    const v16bf qa1 = load_a_bf16(qp + 32, H, lane);

    const int c0 = wave * 256;
    v8f sc[16];
#pragma unroll
    for (int t = 0; t < 16; ++t) {
        const int n0 = c0 + t * 16;
        const __bf16* kp = Kb + ((size_t)(b * S + n0)) * H + h * HD;
        v8f c = {};
        c = wmma_bf16(qa0, load_b_bf16(kp, H, lane), c);
        c = wmma_bf16(qa1, load_b_bf16(kp + 32, H, lane), c);
        const float mv = mask[b * S + n0 + (lane & 15)];
#pragma unroll
        for (int v = 0; v < 8; ++v) c[v] += mv;
        sc[t] = c;
    }

    __shared__ float red[8][16];
    const int rowhalf = (lane >> 4) << 3;     // 0 or 8

    // ---- row max ----
    float rmax[8];
#pragma unroll
    for (int v = 0; v < 8; ++v) {
        float m = sc[0][v];
#pragma unroll
        for (int t = 1; t < 16; ++t) m = fmaxf(m, sc[t][v]);
#pragma unroll
        for (int off = 1; off < 16; off <<= 1) m = fmaxf(m, __shfl_xor(m, off));
        rmax[v] = m;
    }
    if ((lane & 15) == 0) {
#pragma unroll
        for (int v = 0; v < 8; ++v) red[wave][rowhalf + v] = rmax[v];
    }
    __syncthreads();
#pragma unroll
    for (int v = 0; v < 8; ++v) {
        float m = red[0][rowhalf + v];
#pragma unroll
        for (int w = 1; w < 8; ++w) m = fmaxf(m, red[w][rowhalf + v]);
        rmax[v] = m;
    }
    __syncthreads();

    // ---- exp + row sum ----
    float rsum[8];
#pragma unroll
    for (int v = 0; v < 8; ++v) rsum[v] = 0.f;
#pragma unroll
    for (int t = 0; t < 16; ++t)
#pragma unroll
        for (int v = 0; v < 8; ++v) {
            const float e = __expf(sc[t][v] - rmax[v]);
            sc[t][v] = e;
            rsum[v] += e;
        }
#pragma unroll
    for (int v = 0; v < 8; ++v) {
        float s = rsum[v];
#pragma unroll
        for (int off = 1; off < 16; off <<= 1) s += __shfl_xor(s, off);
        rsum[v] = s;
    }
    if ((lane & 15) == 0) {
#pragma unroll
        for (int v = 0; v < 8; ++v) red[wave][rowhalf + v] = rsum[v];
    }
    __syncthreads();
    const float hm = head_mask[h];
#pragma unroll
    for (int v = 0; v < 8; ++v) {
        float s = 0.f;
#pragma unroll
        for (int w = 0; w < 8; ++w) s += red[w][rowhalf + v];
        rsum[v] = hm / s;
    }

    // ---- normalize + store probs ----
    const size_t base = (((size_t)(b * NH + h)) * S + q0) * S;
#pragma unroll
    for (int t = 0; t < 16; ++t) {
        const int n = c0 + t * 16 + (lane & 15);
#pragma unroll
        for (int v = 0; v < 8; ++v) {
            const int m = rowhalf + v;
            probs[base + (size_t)m * S + n] = sc[t][v] * rsum[v];
        }
    }
}

// ---------------------------------------------------------------------------
// ctx[b,s,h*HD+d] = sum_k probs[b,h,s,k] * V[b,k,h,d]   (V given transposed)
// Block: 8 waves x 16 rows = 128 rows; 16 blocks per (b,h).
// Vt tile (64 d-rows x 32 k, shared by all waves) async-staged in LDS.
// ---------------------------------------------------------------------------
__global__ __launch_bounds__(256)
void ctx_kernel(const float* __restrict__ probs, const __bf16* __restrict__ Vt,
                __bf16* __restrict__ ctx) {
    __shared__ __bf16 Bt[2][64 * 32];
    const int mblk = blockIdx.x & 15;
    const int h = (blockIdx.x >> 4) & (NH - 1);
    const int b = blockIdx.x >> 8;
    const int tid  = threadIdx.x;
    const int lane = tid & 31;
    const int wave = tid >> 5;
    const int m0 = mblk * 128 + wave * 16;

    const float*  ap = probs + (((size_t)(b * NH + h)) * S + m0) * S;
    const __bf16* vp = Vt + ((size_t)(b * NH + h)) * HD * S;

    const int srow = tid >> 2;                 // d-row 0..63
    const int soff = (tid & 3) * 8;
    const __bf16* gsrc = vp + (size_t)srow * S + soff;
    const unsigned int l0 = (unsigned int)(size_t)&Bt[0][srow * 32 + soff];
    const unsigned int l1 = (unsigned int)(size_t)&Bt[1][srow * 32 + soff];

    async_copy_b128(l0, gsrc);                 // k-tile 0 -> buf 0

    v8f acc[4] = {};
    const int NIT = S / 32;
    for (int i = 0; i < NIT; ++i) {
        const int k = i * 32;
        if (i + 1 < NIT) {
            async_copy_b128((i & 1) ? l0 : l1, gsrc + (k + 32));
            wait_async<1>();
        } else {
            wait_async<0>();
        }
        __syncthreads();

        v16bf a = load_a_f32(ap + k, S, lane);
        const __bf16* bt = Bt[i & 1];
#pragma unroll
        for (int t = 0; t < 4; ++t)
            acc[t] = wmma_bf16(a, load_b_bf16(bt + t * 16 * 32, 32, lane), acc[t]);

        __syncthreads();
    }

    const int rbase = m0 + ((lane >> 4) << 3);
    const int col = lane & 15;
#pragma unroll
    for (int t = 0; t < 4; ++t) {
        const int d = t * 16 + col;
#pragma unroll
        for (int v = 0; v < 8; ++v) {
            const int s = rbase + v;
            ctx[((size_t)(b * S + s)) * H + h * HD + d] = (__bf16)acc[t][v];
        }
    }
}

// ---------------------------------------------------------------------------
// LayerNorm(h + hidden) * g + b   (one block per row, 256 threads, H=1024)
// ---------------------------------------------------------------------------
__global__ __launch_bounds__(256)
void ln_kernel(const float* __restrict__ go, const float* __restrict__ hidden,
               const float* __restrict__ g, const float* __restrict__ beta,
               float* __restrict__ out) {
    const int row = blockIdx.x;
    const float* xr = go + (size_t)row * H;
    const float* hr = hidden + (size_t)row * H;
    float vals[4], lsum = 0.f, lsq = 0.f;
#pragma unroll
    for (int i = 0; i < 4; ++i) {
        const int c = threadIdx.x + i * 256;
        const float x = xr[c] + hr[c];
        vals[i] = x; lsum += x; lsq += x * x;
    }
    __shared__ float s1[8], s2[8];
    for (int off = 1; off < 32; off <<= 1) {
        lsum += __shfl_xor(lsum, off);
        lsq  += __shfl_xor(lsq, off);
    }
    const int lane = threadIdx.x & 31, wave = threadIdx.x >> 5;
    if (lane == 0) { s1[wave] = lsum; s2[wave] = lsq; }
    __syncthreads();
    if (wave == 0) {
        float a = (lane < 8) ? s1[lane] : 0.f;
        float q = (lane < 8) ? s2[lane] : 0.f;
        for (int off = 1; off < 8; off <<= 1) { a += __shfl_xor(a, off); q += __shfl_xor(q, off); }
        if (lane == 0) { s1[0] = a; s2[0] = q; }
    }
    __syncthreads();
    const float mu = s1[0] * (1.f / H);
    const float var = s2[0] * (1.f / H) - mu * mu;
    const float rstd = rsqrtf(var + EPS);
#pragma unroll
    for (int i = 0; i < 4; ++i) {
        const int c = threadIdx.x + i * 256;
        out[(size_t)row * H + c] = (vals[i] - mu) * rstd * g[c] + beta[c];
    }
}

// ---------------------------------------------------------------------------
extern "C" void kernel_launch(void* const* d_in, const int* in_sizes, int n_in,
                              void* d_out, int out_size, void* d_ws, size_t ws_size,
                              hipStream_t stream) {
    const float* hidden = (const float*)d_in[0];
    const float* amask  = (const float*)d_in[1];
    const float* hmask  = (const float*)d_in[2];
    const float* Wq = (const float*)d_in[3];  const float* bq = (const float*)d_in[4];
    const float* Wk = (const float*)d_in[5];  const float* bk = (const float*)d_in[6];
    const float* Wv = (const float*)d_in[7];  const float* bv = (const float*)d_in[8];
    const float* Wd = (const float*)d_in[9];  const float* bd = (const float*)d_in[10];
    const float* lng = (const float*)d_in[11]; const float* lnb = (const float*)d_in[12];

    float* outp  = (float*)d_out;                       // [B,S,H]
    float* probs = outp + (size_t)MROWS * H;            // [B,NH,S,S]

    // workspace layout (bf16 staging + fp32 pre-LN buffer), ~64 MB total
    __bf16* Xb  = (__bf16*)d_ws;                        // [MROWS,H]
    __bf16* Wqb = Xb  + (size_t)MROWS * H;
    __bf16* Wkb = Wqb + (size_t)H * H;
    __bf16* Wvb = Wkb + (size_t)H * H;
    __bf16* Wdb = Wvb + (size_t)H * H;
    __bf16* Qb  = Wdb + (size_t)H * H;                  // [MROWS,H], pre-scaled
    __bf16* Kb  = Qb  + (size_t)MROWS * H;
    __bf16* Vtb = Kb  + (size_t)MROWS * H;              // [B,NH,HD,S]
    __bf16* Ctx = Vtb + (size_t)MROWS * H;              // [MROWS,H]
    float*  Hf  = (float*)(Ctx + (size_t)MROWS * H);    // [MROWS,H] fp32

    // 1) fp32 -> bf16 staging
    cvt_bf16_kernel<<<(MROWS * H + 255) / 256, 256, 0, stream>>>(hidden, Xb, MROWS * H);
    cvt_bf16_kernel<<<(H * H + 255) / 256, 256, 0, stream>>>(Wq, Wqb, H * H);
    cvt_bf16_kernel<<<(H * H + 255) / 256, 256, 0, stream>>>(Wk, Wkb, H * H);
    cvt_bf16_kernel<<<(H * H + 255) / 256, 256, 0, stream>>>(Wv, Wvb, H * H);
    cvt_bf16_kernel<<<(H * H + 255) / 256, 256, 0, stream>>>(Wd, Wdb, H * H);

    // 2) QKV projections (WMMA + async-LDS staging). Q pre-scaled by 1/sqrt(HD);
    //    V stored transposed [B,NH,HD,S].
    dim3 gg(MROWS / 128, H / 64);
    gemm_bf16_kernel<1><<<gg, 256, 0, stream>>>(Xb, Wqb, bq, Qb, MROWS, H, H, 0.125f);
    gemm_bf16_kernel<1><<<gg, 256, 0, stream>>>(Xb, Wkb, bk, Kb, MROWS, H, H, 1.0f);
    gemm_bf16_kernel<3><<<gg, 256, 0, stream>>>(Xb, Wvb, bv, Vtb, MROWS, H, H, 1.0f);

    // 3) scores + softmax -> probs (second output)
    scores_softmax_kernel<<<B * NH * (S / 16), 256, 0, stream>>>(Qb, Kb, amask, hmask, probs);

    // 4) ctx = probs @ V (WMMA + async-LDS staging)
    ctx_kernel<<<B * NH * (S / 128), 256, 0, stream>>>(probs, Vtb, Ctx);

    // 5) dense projection (WMMA) -> fp32, then residual + LayerNorm
    gemm_bf16_kernel<0><<<gg, 256, 0, stream>>>(Ctx, Wdb, bd, Hf, MROWS, H, H, 1.0f);
    ln_kernel<<<MROWS, 256, 0, stream>>>(Hf, hidden, lng, lnb, outp);
}
